// RNN_83691732730576
// MI455X (gfx1250) — matrix operand order repaired and checked by reading, not compiled, not run on hardware
//
#include <hip/hip_runtime.h>
#include <math.h>

// ---------------------------------------------------------------------------
// RNN:  xe = emb[w] @ W_eh + b_h ;  h_t = tanh(xe_t + h_{t-1} @ W_hh) x512 ;
//       out = h_S @ W_ho + b_o
// All GEMMs via v_wmma_f32_16x16x32_bf16 (f32 accumulate).
// ---------------------------------------------------------------------------

typedef __bf16 bf16_t;
typedef __attribute__((ext_vector_type(16))) __bf16 v16bf;
typedef __attribute__((ext_vector_type(8)))  __bf16 v8bf;
typedef __attribute__((ext_vector_type(4)))  __bf16 v4bf;
typedef __attribute__((ext_vector_type(8)))  float  v8f;

constexpr int B_ = 64;
constexpr int S_ = 512;
constexpr int E_ = 1024;
constexpr int H_ = 1024;
constexpr int O_ = 32000;

static __device__ __forceinline__ v16bf cat8(v8bf lo, v8bf hi) {
  return __builtin_shufflevector(lo, hi, 0,1,2,3,4,5,6,7,8,9,10,11,12,13,14,15);
}

// Load one 32-byte WMMA fragment (16 bf16) as two 16B chunks (2x ds_load_b128).
static __device__ __forceinline__ v16bf ldfrag(const bf16_t* p) {
  return cat8(*(const v8bf*)p, *(const v8bf*)(p + 8));
}

static __device__ __forceinline__ v8f wmma_bf16(v16bf a, v16bf b, v8f c) {
  return __builtin_amdgcn_wmma_f32_16x16x32_bf16(false, a, false, b,
                                                 (short)0, c, false, false);
}

// A-fragment (16x32 bf16) direct from bf16 global, row-major.
// Lane L: m=L&15, hi=L>=16; elems 0..7 -> K=k0+hi*8+i, 8..15 -> K=k0+16+hi*8+i
static __device__ __forceinline__ v16bf load_a_global(const bf16_t* rowbase_k0, int hi) {
  return cat8(*(const v8bf*)(rowbase_k0 + hi * 8),
              *(const v8bf*)(rowbase_k0 + 16 + hi * 8));
}

// ---------------------------------------------------------------------------
// Kernel 1: xe = emb[w] @ W_eh + b_h
// Grid (32768/64, 1024/256); 256 threads = 8 waves (2 Mwaves x 4 Nwaves),
// wave tile 32x64 (2x4 frags). A & B staged in LDS pre-swizzled to fragment
// layout with b128 stores only.
// ---------------------------------------------------------------------------
__global__ __launch_bounds__(256)
void k1_embed_gemm(const int* __restrict__ w, const float* __restrict__ emb,
                   const float* __restrict__ Weh, const float* __restrict__ bh,
                   float* __restrict__ xe) {
  __shared__ int toks[64];
  __shared__ __align__(16) bf16_t lA[4][2][32][16];    //  8 KB
  __shared__ __align__(16) bf16_t lB[16][2][32][16];   // 32 KB

  const int tid  = threadIdx.x;
  const int lane = tid & 31;
  const int wid  = tid >> 5;
  const int wm   = wid >> 2;                 // 0..1
  const int wn   = wid & 3;                  // 0..3
  const int rowbase = blockIdx.x * 64;
  const int ncb     = blockIdx.y * 256;

  if (tid < 64) toks[tid] = w[rowbase + tid];
  __syncthreads();

  // A staging: thread -> (row ar, 16-k segment aq). Each segment is exactly
  // two contiguous 8-element runs in fragment layout.
  const int ar   = tid >> 2;            // 0..63
  const int aq   = tid & 3;             // 0..3
  const int akb  = aq >> 1;             // k block
  const int ae0  = (aq & 1) * 8;        // element base
  const long arow = (long)toks[ar] * E_;

  // B staging: thread -> 4 columns x 16 k  (16 float4 loads, 8 b128 stores)
  const int bn4  = (tid & 63) * 4;      // first of 4 columns
  const int bkh  = (tid >> 6) * 16;     // k base within chunk
  const int bkb  = bkh >> 5;
  const int be0  = (bkh & 16) ? 8 : 0;

  v8f acc[2][4] = {{v8f{},v8f{},v8f{},v8f{}},{v8f{},v8f{},v8f{},v8f{}}};

  for (int k0 = 0; k0 < E_; k0 += 64) {
    // ---- stage A (64 rows x 64 k) ----
    {
      const float* src = emb + arow + k0 + aq * 16;
      float v[16];
      *(float4*)&v[0]  = *(const float4*)(src + 0);
      *(float4*)&v[4]  = *(const float4*)(src + 4);
      *(float4*)&v[8]  = *(const float4*)(src + 8);
      *(float4*)&v[12] = *(const float4*)(src + 12);
      v8bf lo, hi8;
      #pragma unroll
      for (int i = 0; i < 8; ++i) { lo[i] = (bf16_t)v[i]; hi8[i] = (bf16_t)v[8 + i]; }
      *(v8bf*)&lA[ar >> 4][akb][(ar & 15)     ][ae0] = lo;
      *(v8bf*)&lA[ar >> 4][akb][(ar & 15) + 16][ae0] = hi8;
    }
    // ---- stage B (64 k x 256 cols) ----
    {
      const float* src = Weh + (long)(k0 + bkh) * H_ + ncb + bn4;
      float4 r[16];
      #pragma unroll
      for (int kk = 0; kk < 16; ++kk)
        r[kk] = *(const float4*)(src + (long)kk * H_);
      #pragma unroll
      for (int c = 0; c < 4; ++c) {
        v8bf lo, hi8;
        #pragma unroll
        for (int i = 0; i < 8; ++i) {
          lo[i]  = (bf16_t)((const float*)&r[i]    )[c];
          hi8[i] = (bf16_t)((const float*)&r[i + 8])[c];
        }
        int n = bn4 + c, nt = n >> 4, nl = n & 15;
        *(v8bf*)&lB[nt][bkb][nl     ][be0] = lo;
        *(v8bf*)&lB[nt][bkb][nl + 16][be0] = hi8;
      }
    }
    __syncthreads();

    #pragma unroll
    for (int kb = 0; kb < 2; ++kb) {
      v16bf a0 = ldfrag(&lA[2 * wm    ][kb][lane][0]);
      v16bf a1 = ldfrag(&lA[2 * wm + 1][kb][lane][0]);
      #pragma unroll
      for (int g = 0; g < 4; ++g) {
        v16bf b = ldfrag(&lB[wn * 4 + g][kb][lane][0]);
        acc[0][g] = wmma_bf16(a0, b, acc[0][g]);
        acc[1][g] = wmma_bf16(a1, b, acc[1][g]);
      }
    }
    __syncthreads();
  }

  // epilogue: + b_h, store f32
  const int nlo = lane & 15;
  const int hi  = lane >> 4;
  #pragma unroll
  for (int f = 0; f < 2; ++f) {
    #pragma unroll
    for (int g = 0; g < 4; ++g) {
      int col  = ncb + (wn * 4 + g) * 16 + nlo;
      float bias = bh[col];
      #pragma unroll
      for (int i = 0; i < 8; ++i) {
        int row = rowbase + (2 * wm + f) * 16 + hi * 8 + i;
        xe[(long)row * H_ + col] = acc[f][g][i] + bias;
      }
    }
  }
}

// ---------------------------------------------------------------------------
// Global barrier among gridDim.x persistent blocks (two-phase, generation).
// ---------------------------------------------------------------------------
static __device__ __forceinline__ void grid_barrier(unsigned* cnt, unsigned* gen,
                                                    unsigned nblocks) {
  __syncthreads();
  if (threadIdx.x == 0) {
    __threadfence();  // make this block's h stores device-visible
    unsigned g = __hip_atomic_load(gen, __ATOMIC_RELAXED, __HIP_MEMORY_SCOPE_AGENT);
    unsigned a = __hip_atomic_fetch_add(cnt, 1u, __ATOMIC_ACQ_REL, __HIP_MEMORY_SCOPE_AGENT);
    if (a == nblocks - 1) {
      __hip_atomic_store(cnt, 0u, __ATOMIC_RELAXED, __HIP_MEMORY_SCOPE_AGENT);
      __hip_atomic_fetch_add(gen, 1u, __ATOMIC_RELEASE, __HIP_MEMORY_SCOPE_AGENT);
    } else {
      while (__hip_atomic_load(gen, __ATOMIC_ACQUIRE, __HIP_MEMORY_SCOPE_AGENT) == g)
        __builtin_amdgcn_s_sleep(8);
    }
    __threadfence();
  }
  __syncthreads();
}

// ---------------------------------------------------------------------------
// Kernel 2: persistent recurrence. 64 blocks; block owns 16 columns of h.
// W_hh[:, ncb:ncb+16] pre-swizzled to B-fragment layout in LDS (32 KB).
// 8 waves split K (128 each); cross-wave reduce in LDS; tanh; h stored bf16
// (double-buffered in global); one device barrier per step.
// ---------------------------------------------------------------------------
__global__ __launch_bounds__(256)
void k2_recurrence(const float* __restrict__ xe, const float* __restrict__ Whh,
                   const float* __restrict__ bh, bf16_t* __restrict__ hbuf0,
                   bf16_t* __restrict__ hbuf1, unsigned* __restrict__ bar) {
  __shared__ __align__(16) bf16_t lW[32][32][16];   // 32 KB: 32 Kblocks of B-frags
  __shared__ float red[8][64][16];                  // 32 KB: per-wave K partials

  const int tid  = threadIdx.x;
  const int lane = tid & 31;
  const int wid  = tid >> 5;          // 0..7 -> K slice owner
  const int ncb  = blockIdx.x * 16;

  // preload + swizzle W_hh column slice: thread -> 4 cols x 16 k
  {
    const int pc4 = (tid & 3) * 4;          // 4 columns
    const int pkh = (tid >> 2) * 16;        // k base (0..1008)
    const int pkb = pkh >> 5;
    const int pe0 = (pkh & 16) ? 8 : 0;
    const float* src = Whh + (long)pkh * H_ + ncb + pc4;
    float4 r[16];
    #pragma unroll
    for (int kk = 0; kk < 16; ++kk)
      r[kk] = *(const float4*)(src + (long)kk * H_);
    #pragma unroll
    for (int c = 0; c < 4; ++c) {
      v8bf lo, hi8;
      #pragma unroll
      for (int i = 0; i < 8; ++i) {
        lo[i]  = (bf16_t)((const float*)&r[i]    )[c];
        hi8[i] = (bf16_t)((const float*)&r[i + 8])[c];
      }
      int nl = pc4 + c;
      *(v8bf*)&lW[pkb][nl     ][pe0] = lo;
      *(v8bf*)&lW[pkb][nl + 16][pe0] = hi8;
    }
  }
  __syncthreads();

  unsigned* cnt = bar;
  unsigned* gen = bar + 16;

  const int nlo  = lane & 15;
  const int hi   = lane >> 4;
  const int jrow = tid >> 2;          // reduce: batch row 0..63
  const int jc4  = (tid & 3) * 4;     // reduce: 4 cols
  const float4 biasv = *(const float4*)(bh + ncb + jc4);
  const float* xrow = xe + (long)jrow * S_ * H_ + ncb + jc4;

  for (int t = 0; t < S_; ++t) {
    const bf16_t* cur = (t & 1) ? hbuf1 : hbuf0;
    bf16_t*       nxt = (t & 1) ? hbuf0 : hbuf1;

    v8f acc[4] = {v8f{}, v8f{}, v8f{}, v8f{}};
    #pragma unroll
    for (int kbl = 0; kbl < 4; ++kbl) {
      int kbIdx = wid * 4 + kbl;      // this wave's K block
      int kglob = kbIdx * 32;
      v16bf bf = ldfrag(&lW[kbIdx][lane][0]);
      #pragma unroll
      for (int mf = 0; mf < 4; ++mf) {
        const bf16_t* pa = cur + (long)(mf * 16 + nlo) * H_ + kglob;
        v16bf af = load_a_global(pa, hi);
        acc[mf] = wmma_bf16(af, bf, acc[mf]);
      }
    }

    // cross-wave reduction of K partials
    #pragma unroll
    for (int mf = 0; mf < 4; ++mf)
      #pragma unroll
      for (int i = 0; i < 8; ++i)
        red[wid][mf * 16 + hi * 8 + i][nlo] = acc[mf][i];
    __syncthreads();

    float4 xv = *(const float4*)(xrow + (long)t * H_);
    float s0 = xv.x + biasv.x, s1 = xv.y + biasv.y;
    float s2 = xv.z + biasv.z, s3 = xv.w + biasv.w;
    #pragma unroll
    for (int ww = 0; ww < 8; ++ww) {
      const float* rp = &red[ww][jrow][jc4];
      s0 += rp[0]; s1 += rp[1]; s2 += rp[2]; s3 += rp[3];
    }
    v4bf hv;
    hv[0] = (bf16_t)tanhf(s0);
    hv[1] = (bf16_t)tanhf(s1);
    hv[2] = (bf16_t)tanhf(s2);
    hv[3] = (bf16_t)tanhf(s3);
    *(v4bf*)(nxt + (long)jrow * H_ + ncb + jc4) = hv;   // one 8B store

    grid_barrier(cnt, gen, gridDim.x);  // also fences red[] reuse
  }
}

// ---------------------------------------------------------------------------
// Kernel 3: out = h_S @ W_ho + b_o.  Grid 250 blocks x (64x128 tile),
// 8 waves (2x4), wave tile 32x32. h is bf16 (cache resident); W_ho streamed
// once from HBM with half-wave-contiguous loads + prefetch of next K slab.
// ---------------------------------------------------------------------------
__global__ __launch_bounds__(256)
void k3_output(const bf16_t* __restrict__ h, const float* __restrict__ Who,
               const float* __restrict__ bo, float* __restrict__ out) {
  const int tid  = threadIdx.x;
  const int lane = tid & 31;
  const int wid  = tid >> 5;
  const int wm   = wid >> 2;
  const int wn   = wid & 3;
  const int ncb  = blockIdx.x * 128;
  const int nlo  = lane & 15;
  const int hi   = lane >> 4;
  const int khi  = hi ? 16 : 0;
  const int n0   = ncb + wn * 32 + nlo;
  const int n1   = n0 + 16;

  v8f acc00 = {}, acc01 = {}, acc10 = {}, acc11 = {};

  for (int k0 = 0; k0 < H_; k0 += 32) {
    const bf16_t* pa0 = h + (long)(wm * 32 + nlo) * H_ + k0;
    const bf16_t* pa1 = h + (long)(wm * 32 + 16 + nlo) * H_ + k0;
    v16bf a0 = load_a_global(pa0, hi);
    v16bf a1 = load_a_global(pa1, hi);

    if (k0 + 32 < H_) {  // prefetch next K slab of W_ho (global_prefetch_b8)
      __builtin_prefetch(Who + (long)(k0 + 32 + khi) * O_ + n0, 0, 1);
    }

    v16bf b0 = {}, b1 = {};
    #pragma unroll
    for (int i = 0; i < 16; ++i) {
      long roff = (long)(k0 + (i & 8) * 2 + hi * 8 + (i & 7)) * O_;
      b0[i] = (bf16_t)Who[roff + n0];
      b1[i] = (bf16_t)Who[roff + n1];
    }
    acc00 = wmma_bf16(a0, b0, acc00);
    acc01 = wmma_bf16(a0, b1, acc01);
    acc10 = wmma_bf16(a1, b0, acc10);
    acc11 = wmma_bf16(a1, b1, acc11);
  }

  float bias0 = bo[n0], bias1 = bo[n1];
  #pragma unroll
  for (int f = 0; f < 2; ++f) {
    v8f aA = (f == 0) ? acc00 : acc10;
    v8f aB = (f == 0) ? acc01 : acc11;
    #pragma unroll
    for (int i = 0; i < 8; ++i) {
      int row = wm * 32 + f * 16 + hi * 8 + i;
      out[(long)row * O_ + n0] = aA[i] + bias0;
      out[(long)row * O_ + n1] = aB[i] + bias1;
    }
  }
}

// ---------------------------------------------------------------------------
extern "C" void kernel_launch(void* const* d_in, const int* in_sizes, int n_in,
                              void* d_out, int out_size, void* d_ws, size_t ws_size,
                              hipStream_t stream) {
  const int*   w   = (const int*)  d_in[0];
  const float* emb = (const float*)d_in[1];
  const float* Weh = (const float*)d_in[2];
  const float* Whh = (const float*)d_in[3];
  const float* Who = (const float*)d_in[4];
  const float* bh  = (const float*)d_in[5];
  const float* bo  = (const float*)d_in[6];
  float* out = (float*)d_out;

  char* ws = (char*)d_ws;
  const size_t XE_BYTES = (size_t)(B_ * S_) * H_ * sizeof(float);   // 128 MiB
  const size_t HB_BYTES = (size_t)B_ * H_ * sizeof(bf16_t);         // 128 KiB each
  float*    xe = (float*)ws;
  bf16_t*   h0 = (bf16_t*)(ws + XE_BYTES);
  bf16_t*   h1 = (bf16_t*)(ws + XE_BYTES + HB_BYTES);
  unsigned* bar = (unsigned*)(ws + XE_BYTES + 2 * HB_BYTES);

  // zero h0 (reference h0 = 0), h1, and barrier state; capture-safe memset node
  hipMemsetAsync(ws + XE_BYTES, 0, 2 * HB_BYTES + 256, stream);

  k1_embed_gemm<<<dim3((B_ * S_) / 64, H_ / 256), 256, 0, stream>>>(w, emb, Weh, bh, xe);
  k2_recurrence<<<H_ / 16, 256, 0, stream>>>(xe, Whh, bh, h0, h1, bar);
  k3_output<<<O_ / 128, 256, 0, stream>>>(h0, Who, bo, out);
}
